// ResGCN_11596411699465
// MI455X (gfx1250) — matrix operand order repaired and checked by reading, not compiled, not run on hardware
//
#include <hip/hip_runtime.h>
#include <hip/hip_bf16.h>

// ---------------- CDNA5 WMMA types ----------------
typedef __attribute__((ext_vector_type(16))) __bf16 v16bf;
typedef __attribute__((ext_vector_type(8)))  __bf16 v8bf;
typedef __attribute__((ext_vector_type(8)))  float  v8f;

#define HOST_N 200000
#define HOST_E 3200000
#define HOST_G 2000

// ---------------- small utility kernels ----------------
__global__ void fill_kernel(float* __restrict__ p, long long n, float v) {
  long long i = (long long)blockIdx.x * blockDim.x + threadIdx.x;
  if (i < n) p[i] = v;
}

__global__ void copy_kernel(float* __restrict__ dstp, const float* __restrict__ srcp, long long n) {
  long long i = (long long)blockIdx.x * blockDim.x + threadIdx.x;
  if (i < n) dstp[i] = srcp[i];
}

// ---------------- degree / norm ----------------
__global__ void degree_kernel(float* __restrict__ deg, const long long* __restrict__ dst, int E) {
  int e = blockIdx.x * blockDim.x + threadIdx.x;
  if (e < E) atomicAdd(&deg[dst[e]], 1.0f);
}

__global__ void norm_kernel(float* __restrict__ nv, int n) {
  int i = blockIdx.x * blockDim.x + threadIdx.x;
  if (i >= n) return;
  float d = nv[i];
  nv[i] = (d > 0.0f) ? rsqrtf(fmaxf(d, 1.0f)) : 0.0f;
}

// ---------------- edge scatter: p[dst] += h[src]*norm[src] (4 feats/thread) ----------------
__global__ void edge_scatter(float* __restrict__ p, const float* __restrict__ hsrc,
                             const float* __restrict__ nv,
                             const long long* __restrict__ src, const long long* __restrict__ dst,
                             int E) {
  int e = blockIdx.x * blockDim.x + threadIdx.x;
  if (e >= E) return;
  int f = blockIdx.y << 2;
  long long s = src[e], d = dst[e];
  float ns = nv[s];
  const float4 hv = *(const float4*)(hsrc + s * 128 + f);
  float* pp = p + d * 128 + f;
  atomicAdd(pp + 0, hv.x * ns);
  atomicAdd(pp + 1, hv.y * ns);
  atomicAdd(pp + 2, hv.z * ns);
  atomicAdd(pp + 3, hv.w * ns);
}

// p[i,f] *= norm[i]  (d == 128)
__global__ void row_scale(float* __restrict__ p, const float* __restrict__ nv, long long total) {
  long long idx = (long long)blockIdx.x * blockDim.x + threadIdx.x;
  if (idx >= total) return;
  p[idx] *= nv[idx >> 7];
}

// ---------------- concat [h | p1 | p2] -> bf16 [N,384] ----------------
__global__ void cast_concat(__bf16* __restrict__ Hcat, const float* __restrict__ h,
                            const float* __restrict__ p1, const float* __restrict__ p2,
                            long long total) {
  long long idx = (long long)blockIdx.x * blockDim.x + threadIdx.x;
  if (idx >= total) return;
  long long i = idx / 384;
  int c = (int)(idx - i * 384);
  float v;
  if (c < 128)       v = h [i * 128 + c];
  else if (c < 256)  v = p1[i * 128 + c - 128];
  else               v = p2[i * 128 + c - 256];
  Hcat[idx] = (__bf16)v;
}

// W [384, dout] f32  ->  Wt [dout, 384] bf16 (transposed for contiguous B-fragment loads)
__global__ void cast_w(__bf16* __restrict__ Wt, const float* __restrict__ W, int dout) {
  int idx = blockIdx.x * blockDim.x + threadIdx.x;
  if (idx >= 384 * dout) return;
  int n = idx / 384, k = idx - n * 384;
  Wt[idx] = (__bf16)W[(long long)k * dout + n];
}

// ---------------- WMMA GEMM: C[M,NDIM] = A[M,384] @ Bt[NDIM,384]^T ----------------
// One wave -> one 16-row stripe, all NDIM/16 column tiles (A fragment reused).
template<int NDIM>
__global__ void wmma_gemm(const __bf16* __restrict__ A, const __bf16* __restrict__ Bt,
                          float* __restrict__ C, int M) {
  constexpr int KDIM = 384;
  constexpr int NT = NDIM / 16;
  int wave = threadIdx.x >> 5;
  int lane = threadIdx.x & 31;
  int mt = blockIdx.x * (blockDim.x >> 5) + wave;
  if (mt * 16 >= M) return;
  int lo = lane & 15;
  int hi = lane >> 4;

  const __bf16* arow = A + (long long)(mt * 16 + lo) * KDIM;
  v8f acc[NT] = {};

#pragma unroll
  for (int k0 = 0; k0 < KDIM; k0 += 32) {
    // A fragment (16x32 bf16): lanes 0-15 hold K in k0..k0+15,
    // lanes 16-31 hold K in k0+16..k0+31 (matches B fragment order)
    int ka = k0 + hi * 16;
    v16bf a = *(const v16bf*)(arow + ka);
    __builtin_prefetch(arow + ka + 64, 0, 1);   // global_prefetch for next K slice

#pragma unroll
    for (int nt = 0; nt < NT; ++nt) {
      // B fragment (32x16): N = lane&15, lanes 0-15 K=k0..k0+15, lanes 16-31 K=k0+16..k0+31
      const __bf16* bcol = Bt + (long long)(nt * 16 + lo) * KDIM + k0 + hi * 16;
      v16bf b = *(const v16bf*)bcol;
      acc[nt] = __builtin_amdgcn_wmma_f32_16x16x32_bf16(
          false, a, false, b, (short)0, acc[nt], false, false);
    }
  }

#pragma unroll
  for (int nt = 0; nt < NT; ++nt) {
#pragma unroll
    for (int r = 0; r < 8; ++r) {
      // C/D layout: VGPR r -> M = r + 8*hi, N = lane&15
      C[(long long)(mt * 16 + r + hi * 8) * NDIM + nt * 16 + lo] = acc[nt][r];
    }
  }
}

// ---------------- batchnorm stats / apply ----------------
__global__ void bn_stats(const float* __restrict__ g, long long rows, int d,
                         float* __restrict__ sums, float* __restrict__ sumsq) {
  int f = threadIdx.x;                 // blockDim.x == d
  long long r0 = (long long)blockIdx.x * 512;
  long long r1 = r0 + 512 < rows ? r0 + 512 : rows;
  float s = 0.f, s2 = 0.f;
  for (long long r = r0; r < r1; ++r) {
    float v = g[r * d + f];
    s += v; s2 += v * v;
  }
  atomicAdd(&sums[f], s);
  atomicAdd(&sumsq[f], s2);
}

__global__ void bn_apply(const float* __restrict__ gin, float* __restrict__ hout,
                         float* __restrict__ skipbuf,
                         const float* __restrict__ sums, const float* __restrict__ sumsq,
                         const float* __restrict__ gamma, const float* __restrict__ beta,
                         long long total, int d, float invN, int use_skip, int store_skip) {
  long long idx = (long long)blockIdx.x * blockDim.x + threadIdx.x;
  if (idx >= total) return;
  int f = (int)(idx % d);
  float mean = sums[f] * invN;
  float var  = sumsq[f] * invN - mean * mean;
  float v = (gin[idx] - mean) * rsqrtf(var + 1e-5f) * gamma[f] + beta[f];
  v = fmaxf(v, 0.0f);
  if (use_skip)  v += skipbuf[idx];
  hout[idx] = v;
  if (store_skip) skipbuf[idx] = v;
}

// ---------------- pooling ----------------
__device__ inline void atomicMaxF(float* addr, float val) {
  int old = __float_as_int(*addr);
  while (__int_as_float(old) < val) {
    int prev = atomicCAS((int*)addr, old, __float_as_int(val));
    if (prev == old) break;
    old = prev;
  }
}
__device__ inline void atomicMinF(float* addr, float val) {
  int old = __float_as_int(*addr);
  while (__int_as_float(old) > val) {
    int prev = atomicCAS((int*)addr, old, __float_as_int(val));
    if (prev == old) break;
    old = prev;
  }
}

__global__ void count_kernel(float* __restrict__ pcount, const long long* __restrict__ seg, int n) {
  int i = blockIdx.x * blockDim.x + threadIdx.x;
  if (i < n) atomicAdd(&pcount[seg[i]], 1.0f);
}

__global__ void pool_kernel(const float* __restrict__ h, const long long* __restrict__ seg,
                            float* __restrict__ psum, float* __restrict__ psumsq,
                            float* __restrict__ pmax, float* __restrict__ pmin, long long total) {
  long long idx = (long long)blockIdx.x * blockDim.x + threadIdx.x;
  if (idx >= total) return;
  long long i = idx >> 6;
  int f = (int)(idx & 63);
  long long g = seg[i];
  float v = h[i * 64 + f];
  long long o = g * 64 + f;
  atomicAdd(&psum[o], v);
  atomicAdd(&psumsq[o], v * v);
  atomicMaxF(&pmax[o], v);
  atomicMinF(&pmin[o], v);
}

__global__ void pooled_kernel(const float* __restrict__ psum, const float* __restrict__ psumsq,
                              const float* __restrict__ pmax, const float* __restrict__ pmin,
                              const float* __restrict__ pcount, float* __restrict__ pooled,
                              float* __restrict__ chsum, float* __restrict__ chsumsq, int total) {
  __shared__ float sacc[8];
  if (threadIdx.x < 8) sacc[threadIdx.x] = 0.0f;
  __syncthreads();
  int idx = blockIdx.x * blockDim.x + threadIdx.x;
  if (idx < total) {
    int g = idx >> 6, f = idx & 63;
    float c = pcount[g];
    float s = psum[idx];
    float mean = s / fmaxf(c, 1.0f);
    float var = (psumsq[idx] - c * mean * mean) / fmaxf(c - 1.0f, 1.0f);
    float sd = sqrtf(fmaxf(var, 0.0f));
    float vals[4] = { pmax[idx], pmin[idx], mean, sd };
#pragma unroll
    for (int c4 = 0; c4 < 4; ++c4) {
      pooled[((long long)g * 4 + c4) * 64 + f] = vals[c4];
      atomicAdd(&sacc[c4], vals[c4]);
      atomicAdd(&sacc[4 + c4], vals[c4] * vals[c4]);
    }
  }
  __syncthreads();
  if (threadIdx.x < 4)       atomicAdd(&chsum[threadIdx.x], sacc[threadIdx.x]);
  else if (threadIdx.x < 8)  atomicAdd(&chsumsq[threadIdx.x - 4], sacc[threadIdx.x]);
}

// ---------------- aggregation BN + Linear(256->64) + Linear(64->1) ----------------
__global__ void agg_final(const float* __restrict__ pooled,
                          const float* __restrict__ chsum, const float* __restrict__ chsumsq,
                          const float* __restrict__ agg_gamma, const float* __restrict__ agg_beta,
                          const float* __restrict__ agg_W, const float* __restrict__ agg_b,
                          const float* __restrict__ out_W, const float* __restrict__ out_b,
                          float* __restrict__ outp, int statN) {
  int g = blockIdx.x;
  int t = threadIdx.x;   // 64 threads
  __shared__ float flat[256];
  __shared__ float red[64];
  float inv = 1.0f / (float)statN;
#pragma unroll
  for (int c = 0; c < 4; ++c) {
    float m = chsum[c] * inv;
    float v = chsumsq[c] * inv - m * m;
    float p = pooled[((long long)g * 4 + c) * 64 + t];
    flat[c * 64 + t] = (p - m) * rsqrtf(v + 1e-5f) * agg_gamma[c] + agg_beta[c];
  }
  __syncthreads();
  float acc = agg_b[t];
#pragma unroll 4
  for (int j = 0; j < 256; ++j) acc += flat[j] * agg_W[t * 256 + j];
  red[t] = acc * out_W[t];
  __syncthreads();
  for (int s = 32; s > 0; s >>= 1) {
    if (t < s) red[t] += red[t + s];
    __syncthreads();
  }
  if (t == 0) outp[g] = red[0] + out_b[0];
}

// ---------------- host orchestration ----------------
extern "C" void kernel_launch(void* const* d_in, const int* in_sizes, int n_in,
                              void* d_out, int out_size, void* d_ws, size_t ws_size,
                              hipStream_t stream) {
  (void)in_sizes; (void)n_in; (void)out_size;
  const int N = HOST_N, E = HOST_E, G = HOST_G;

  const float*     x            = (const float*)d_in[0];
  const long long* eidx         = (const long long*)d_in[1];
  const long long* src          = eidx;
  const long long* dst          = eidx + E;
  const long long* seg          = (const long long*)d_in[2];
  const float*     conv_W       = (const float*)d_in[3];
  const float*     conv_W_last  = (const float*)d_in[4];
  const float*     bn_gamma     = (const float*)d_in[5];
  const float*     bn_beta      = (const float*)d_in[6];
  const float*     bn_gamma_l   = (const float*)d_in[7];
  const float*     bn_beta_l    = (const float*)d_in[8];
  const float*     agg_gamma    = (const float*)d_in[9];
  const float*     agg_beta     = (const float*)d_in[10];
  const float*     agg_W        = (const float*)d_in[11];
  const float*     agg_b        = (const float*)d_in[12];
  const float*     out_W        = (const float*)d_in[13];
  const float*     out_b        = (const float*)d_in[14];
  float*           out          = (float*)d_out;

  char* w = (char*)d_ws;
  size_t off = 0;
  auto carve = [&](size_t bytes) -> char* {
    char* p = w + off;
    off = (off + bytes + 255) & ~(size_t)255;
    return p;
  };
  float*  normv  = (float*) carve((size_t)N * 4);
  float*  h      = (float*) carve((size_t)N * 128 * 4);
  float*  p1     = (float*) carve((size_t)N * 128 * 4);
  float*  p2     = (float*) carve((size_t)N * 128 * 4);   // doubles as GEMM output
  float*  skipA  = (float*) carve((size_t)N * 128 * 4);
  float*  skipB  = (float*) carve((size_t)N * 128 * 4);
  __bf16* Hcat   = (__bf16*)carve((size_t)N * 384 * 2);
  __bf16* Wt     = (__bf16*)carve((size_t)384 * 128 * 2);
  float*  sums   = (float*) carve(128 * 4);
  float*  sumsq  = (float*) carve(128 * 4);
  float*  psum   = (float*) carve((size_t)G * 64 * 4);
  float*  psumsq = (float*) carve((size_t)G * 64 * 4);
  float*  pmax   = (float*) carve((size_t)G * 64 * 4);
  float*  pmin   = (float*) carve((size_t)G * 64 * 4);
  float*  pcount = (float*) carve((size_t)G * 4);
  float*  pooled = (float*) carve((size_t)G * 4 * 64 * 4);
  float*  chsum  = (float*) carve(4 * 4);
  float*  chsumsq= (float*) carve(4 * 4);
  if (off > ws_size) return;  // workspace too small: bail deterministically

#define L1D(kern, n, ...) kern<<<dim3((unsigned)(((long long)(n) + 255) / 256)), dim3(256), 0, stream>>>(__VA_ARGS__)

  // degree + norm
  L1D(fill_kernel, N, normv, (long long)N, 0.0f);
  L1D(degree_kernel, E, normv, dst, E);
  L1D(norm_kernel, N, normv, N);
  // h = x
  L1D(copy_kernel, (long long)N * 128, h, x, (long long)N * 128);

  for (int l = 0; l < 6; ++l) {
    const int dout = (l < 5) ? 128 : 64;
    const float* Wl    = (l < 5) ? (conv_W + (size_t)l * 384 * 128) : conv_W_last;
    const float* gamma = (l < 5) ? (bn_gamma + l * 128) : bn_gamma_l;
    const float* beta  = (l < 5) ? (bn_beta  + l * 128) : bn_beta_l;

    // p1 = prop(h) = norm * scatter(h*norm)
    L1D(fill_kernel, (long long)N * 128, p1, (long long)N * 128, 0.0f);
    edge_scatter<<<dim3((E + 255) / 256, 32), dim3(256), 0, stream>>>(p1, h, normv, src, dst, E);
    L1D(row_scale, (long long)N * 128, p1, normv, (long long)N * 128);
    // p2 = prop(p1)
    L1D(fill_kernel, (long long)N * 128, p2, (long long)N * 128, 0.0f);
    edge_scatter<<<dim3((E + 255) / 256, 32), dim3(256), 0, stream>>>(p2, p1, normv, src, dst, E);
    L1D(row_scale, (long long)N * 128, p2, normv, (long long)N * 128);

    // Hcat = bf16 concat([h, p1, p2]);  Wt = bf16 transpose(Wl)
    L1D(cast_concat, (long long)N * 384, Hcat, h, p1, p2, (long long)N * 384);
    L1D(cast_w, 384 * dout, Wt, Wl, dout);

    // GEMM (overwrite p2 with output — stream-ordered after cast_concat consumed it)
    const int mtiles = N / 16;                       // 12500
    const int blocks = (mtiles + 7) / 8;             // 8 waves per block
    if (dout == 128)
      wmma_gemm<128><<<dim3(blocks), dim3(256), 0, stream>>>(Hcat, Wt, p2, N);
    else
      wmma_gemm<64><<<dim3(blocks), dim3(256), 0, stream>>>(Hcat, Wt, p2, N);

    // BN stats + apply (+relu, +skip), writes new h and the skip ring slot
    L1D(fill_kernel, dout, sums, (long long)dout, 0.0f);
    L1D(fill_kernel, dout, sumsq, (long long)dout, 0.0f);
    bn_stats<<<dim3((N + 511) / 512), dim3(dout), 0, stream>>>(p2, (long long)N, dout, sums, sumsq);
    const int use_skip  = (l >= 2 && l < 5) ? 1 : 0;
    const int store_skip = (l < 5) ? 1 : 0;
    float* sbuf = (l % 2 == 0) ? skipA : skipB;      // slot (l-2)%2 == l%2 holds outs[l-2]
    L1D(bn_apply, (long long)N * dout, p2, h, sbuf, sums, sumsq, gamma, beta,
        (long long)N * dout, dout, 1.0f / (float)N, use_skip, store_skip);
  }

  // ---- ragged pooling (h is now [N,64]) ----
  L1D(fill_kernel, (long long)G * 64, psum,   (long long)G * 64, 0.0f);
  L1D(fill_kernel, (long long)G * 64, psumsq, (long long)G * 64, 0.0f);
  L1D(fill_kernel, (long long)G * 64, pmax,   (long long)G * 64, -3.402823466e38f);
  L1D(fill_kernel, (long long)G * 64, pmin,   (long long)G * 64,  3.402823466e38f);
  L1D(fill_kernel, G, pcount, (long long)G, 0.0f);
  L1D(count_kernel, N, pcount, seg, N);
  L1D(pool_kernel, (long long)N * 64, h, seg, psum, psumsq, pmax, pmin, (long long)N * 64);

  L1D(fill_kernel, 4, chsum,   4, 0.0f);
  L1D(fill_kernel, 4, chsumsq, 4, 0.0f);
  L1D(pooled_kernel, G * 64, psum, psumsq, pmax, pmin, pcount, pooled, chsum, chsumsq, G * 64);

  agg_final<<<dim3(G), dim3(64), 0, stream>>>(pooled, chsum, chsumsq, agg_gamma, agg_beta,
                                              agg_W, agg_b, out_W, out_b, out, G * 64);
#undef L1D
}
